// MultiHeadAttention_62199716381091
// MI455X (gfx1250) — matrix operand order repaired and checked by reading, not compiled, not run on hardware
//
#include <hip/hip_runtime.h>

// ---------------------------------------------------------------------------
// MultiHeadAttention forward for MI455X (gfx1250), bf16 WMMA + f32 accumulate.
// B=2, S=2048, E=1024, H=16, Dh=64.
// Pipeline: cvt(f32->bf16) -> QKV proj GEMMs -> flash attention -> out proj.
// ---------------------------------------------------------------------------

typedef __bf16 bf16;
typedef __attribute__((ext_vector_type(16))) __bf16 v16bf;
typedef __attribute__((ext_vector_type(8)))  __bf16 v8bf;
typedef __attribute__((ext_vector_type(8)))  float  v8f;

constexpr int BB  = 2;
constexpr int SS  = 2048;
constexpr int EE  = 1024;
constexpr int HH  = 16;
constexpr int DH  = 64;
constexpr int ROWS = BB * SS;            // 4096

__device__ __forceinline__ v8f wmma_bf16(v16bf a, v16bf b, v8f c) {
    return __builtin_amdgcn_wmma_f32_16x16x32_bf16(
        /*neg_a=*/false, a, /*neg_b=*/false, b,
        /*c_mod=*/(short)0, c, /*reuse_a=*/false, /*reuse_b=*/false);
}

// A-matrix fragment (16x32 bf16) from row-major bf16.
// Lanes 0-15: row M=lane,   K = kk+[0..7] (v0..3) and kk+[16..23] (v4..7)
// Lanes16-31: row M=lane-16,K = kk+[8..15] and kk+[24..31]
__device__ __forceinline__ v16bf afrag_bf16(const bf16* rowp, int kk, int lane) {
    const int kb = (lane & 16) ? 8 : 0;
    v8bf c0 = *(const v8bf*)(rowp + kk + kb);
    v8bf c1 = *(const v8bf*)(rowp + kk + kb + 16);
    v16bf a;
#pragma unroll
    for (int i = 0; i < 8; ++i) { a[i] = c0[i]; a[8 + i] = c1[i]; }
    return a;
}

// B-matrix fragment (32x16 bf16), lane n holds column N=n%16:
// lanes 0-15: K = kk+[0..7],[16..23]; lanes 16-31: K = kk+[8..15],[24..31]
__device__ __forceinline__ v16bf bfrag_bf16(const bf16* colp, int kk, int lane) {
    const int off = kk + ((lane & 16) ? 8 : 0);
    v8bf c0 = *(const v8bf*)(colp + off);
    v8bf c1 = *(const v8bf*)(colp + off + 16);
    v16bf b;
#pragma unroll
    for (int i = 0; i < 8; ++i) { b[i] = c0[i]; b[8 + i] = c1[i]; }
    return b;
}

// ---------------------------------------------------------------------------
// Kernel 0: bulk f32 -> bf16 conversion (grid-stride, 8 elems/thread/iter).
// ---------------------------------------------------------------------------
__global__ void cvt_bf16_kernel(const float* __restrict__ src,
                                bf16* __restrict__ dst, int n8) {
    for (int i = blockIdx.x * blockDim.x + threadIdx.x; i < n8;
         i += gridDim.x * blockDim.x) {
        v8f v = *(const v8f*)(src + (size_t)i * 8);
        v8bf o;
#pragma unroll
        for (int j = 0; j < 8; ++j) o[j] = (bf16)v[j];
        *(v8bf*)(dst + (size_t)i * 8) = o;
    }
}

// ---------------------------------------------------------------------------
// Kernel 1: Q/K/V projections (Xbf @ Wbf^T + bias), bf16 in -> bf16 out.
// grid = (32, 16, 3), block = 128 (4 waves). Wave = 32(M) x 64(N) tile.
// z=0: Q -> [B,H,S,Dh]; z=1: K -> [B,H,S,Dh]; z=2: V -> [B,H,Dh,S] (transposed)
// ---------------------------------------------------------------------------
__global__ void qkv_proj_kernel(const bf16* __restrict__ Xq,
                                const bf16* __restrict__ Xk,
                                const bf16* __restrict__ Xv,
                                const bf16* __restrict__ Wq,
                                const bf16* __restrict__ Wk,
                                const bf16* __restrict__ Wv,
                                const float* __restrict__ bq,
                                const float* __restrict__ bk,
                                const float* __restrict__ bv,
                                bf16* __restrict__ Qh,
                                bf16* __restrict__ Kh,
                                bf16* __restrict__ Vt) {
    const int lane = threadIdx.x & 31;
    const int wave = threadIdx.x >> 5;
    const int mt   = blockIdx.x * 4 + wave;     // 0..127  (32-row tile)
    const int nb   = blockIdx.y * 64;           // output column base
    const int which = blockIdx.z;

    const bf16*  X    = (which == 0) ? Xq : (which == 1) ? Xk : Xv;
    const bf16*  W    = (which == 0) ? Wq : (which == 1) ? Wk : Wv;
    const float* bias = (which == 0) ? bq : (which == 1) ? bk : bv;

    const int n16 = lane & 15;
    const int hi8 = (lane & 16) ? 8 : 0;
    const int row0 = mt * 32;
    const bf16* arow0 = X + (size_t)(row0 + n16) * EE;
    const bf16* arow1 = X + (size_t)(row0 + 16 + n16) * EE;

    v8f acc[2][4] = {};
    for (int kk = 0; kk < EE; kk += 32) {
        v16bf a0 = afrag_bf16(arow0, kk, lane);
        v16bf a1 = afrag_bf16(arow1, kk, lane);
#pragma unroll
        for (int t = 0; t < 4; ++t) {
            const bf16* brow = W + (size_t)(nb + t * 16 + n16) * EE;
            v16bf b = bfrag_bf16(brow, kk, lane);
            acc[0][t] = wmma_bf16(a0, b, acc[0][t]);
            acc[1][t] = wmma_bf16(a1, b, acc[1][t]);
        }
    }

#pragma unroll
    for (int t = 0; t < 4; ++t) {
        const int col = nb + t * 16 + n16;
        const int h = col >> 6, d = col & 63;
        const float bv_ = bias[col];
#pragma unroll
        for (int mi = 0; mi < 2; ++mi) {
#pragma unroll
            for (int r = 0; r < 8; ++r) {
                const int m  = row0 + mi * 16 + r + hi8;
                const int b_ = m >> 11, s = m & (SS - 1);
                const float v = acc[mi][t][r] + bv_;
                if (which == 2) {
                    Vt[((size_t)((b_ * HH + h) * DH + d)) * SS + s] = (bf16)v;
                } else {
                    bf16* O = (which == 0) ? Qh : Kh;
                    O[((size_t)((b_ * HH + h) * SS + s)) * DH + d] = (bf16)v;
                }
            }
        }
    }
}

// ---------------------------------------------------------------------------
// Kernel 2: flash-style attention. One wave per (b,h,q-tile of 16 rows),
// streaming 32 keys per iteration. grid = 1024, block = 128 (4 waves).
// ---------------------------------------------------------------------------
__global__ void attention_kernel(const bf16* __restrict__ Qh,
                                 const bf16* __restrict__ Kh,
                                 const bf16* __restrict__ Vt,
                                 bf16* __restrict__ At) {
    __shared__ __align__(16) bf16 plds[4 * 16 * 32];   // 1KB per wave

    const int lane = threadIdx.x & 31;
    const int wave = threadIdx.x >> 5;
    const int gid  = blockIdx.x * 4 + wave;            // 0..4095
    const int qt   = gid & 127;                        // q-tile in [0,128)
    const int bh   = gid >> 7;                         // b*H + h in [0,32)

    const bf16* Qb = Qh + (size_t)bh * SS * DH;
    const bf16* Kb = Kh + (size_t)bh * SS * DH;
    const bf16* Vb = Vt + (size_t)bh * DH * SS;
    bf16* pw = plds + wave * (16 * 32);

    const int n16 = lane & 15;
    const int hi8 = (lane & 16) ? 8 : 0;

    // Persistent Q fragments (16 rows x 64 = two k32 A-frags)
    const bf16* qrow = Qb + (size_t)(qt * 16 + n16) * DH;
    const v16bf qa0 = afrag_bf16(qrow, 0,  lane);
    const v16bf qa1 = afrag_bf16(qrow, 32, lane);

    float mrow[8], lrow[8];
#pragma unroll
    for (int r = 0; r < 8; ++r) { mrow[r] = -__builtin_inff(); lrow[r] = 0.0f; }
    v8f acc[4] = {};

    const float scale = 0.125f;   // 1/sqrt(64)

    for (int kc = 0; kc < SS; kc += 32) {
        // ---- scores: 16 q x 32 keys = two 16x16 C-tiles, K-dim 64 = 2 steps
        const bf16* krow0 = Kb + (size_t)(kc + n16) * DH;
        const bf16* krow1 = Kb + (size_t)(kc + 16 + n16) * DH;
        v8f s0 = {}, s1 = {};
        s0 = wmma_bf16(qa0, bfrag_bf16(krow0, 0,  lane), s0);
        s0 = wmma_bf16(qa1, bfrag_bf16(krow0, 32, lane), s0);
        s1 = wmma_bf16(qa0, bfrag_bf16(krow1, 0,  lane), s1);
        s1 = wmma_bf16(qa1, bfrag_bf16(krow1, 32, lane), s1);

        // ---- online softmax per row group (rows live across 16-lane halves)
        float p0v[8], p1v[8], alpha[8];
#pragma unroll
        for (int r = 0; r < 8; ++r) {
            float v0 = s0[r] * scale, v1 = s1[r] * scale;
            float mx = fmaxf(v0, v1);
#pragma unroll
            for (int off = 1; off < 16; off <<= 1)
                mx = fmaxf(mx, __shfl_xor(mx, off, 32));
            const float mnew = fmaxf(mrow[r], mx);
            const float a_   = __expf(mrow[r] - mnew);
            const float p0   = __expf(v0 - mnew);
            const float p1   = __expf(v1 - mnew);
            float rs = p0 + p1;
#pragma unroll
            for (int off = 1; off < 16; off <<= 1)
                rs += __shfl_xor(rs, off, 32);
            lrow[r] = lrow[r] * a_ + rs;
            mrow[r] = mnew;
            alpha[r] = a_;
            p0v[r] = p0; p1v[r] = p1;
        }
#pragma unroll
        for (int t = 0; t < 4; ++t)
#pragma unroll
            for (int r = 0; r < 8; ++r) acc[t][r] *= alpha[r];

        // ---- transpose P (C/D layout -> A-frag layout) via per-wave LDS tile
#pragma unroll
        for (int r = 0; r < 8; ++r) {
            const int m = r + hi8;
            pw[m * 32 + n16]      = (bf16)p0v[r];
            pw[m * 32 + 16 + n16] = (bf16)p1v[r];
        }
        asm volatile("s_wait_dscnt 0" ::: "memory");
        const int kb = (lane & 16) ? 8 : 0;
        v8bf c0 = *(const v8bf*)(pw + n16 * 32 + kb);
        v8bf c1 = *(const v8bf*)(pw + n16 * 32 + kb + 16);
        v16bf pa;
#pragma unroll
        for (int i = 0; i < 8; ++i) { pa[i] = c0[i]; pa[8 + i] = c1[i]; }

        // ---- O += P(16x32) @ V(32x64): 4 d-tiles, V^T layout is contiguous
#pragma unroll
        for (int t = 0; t < 4; ++t) {
            const bf16* vrow = Vb + (size_t)(t * 16 + n16) * SS + kc;
            acc[t] = wmma_bf16(pa, bfrag_bf16(vrow, 0, lane), acc[t]);
        }
    }

    // ---- epilogue: normalize and store bf16 attn [rows=B*S, cols=E]
    float inv[8];
#pragma unroll
    for (int r = 0; r < 8; ++r) inv[r] = 1.0f / lrow[r];
    const int b_ = bh >> 4, h_ = bh & 15;
#pragma unroll
    for (int t = 0; t < 4; ++t) {
        const int d = t * 16 + n16;
#pragma unroll
        for (int r = 0; r < 8; ++r) {
            const size_t row = (size_t)b_ * SS + qt * 16 + r + hi8;
            At[row * EE + h_ * 64 + d] = (bf16)(acc[t][r] * inv[r]);
        }
    }
}

// ---------------------------------------------------------------------------
// Kernel 3: output projection attn(bf16) @ Wo^T(bf16) + bo -> fp32 out.
// grid = (32, 16), block = 128 (4 waves). Wave = 32(M) x 64(N) tile.
// ---------------------------------------------------------------------------
__global__ void out_proj_kernel(const bf16* __restrict__ At,
                                const bf16* __restrict__ Wo,
                                const float* __restrict__ bo,
                                float* __restrict__ Out) {
    const int lane = threadIdx.x & 31;
    const int wave = threadIdx.x >> 5;
    const int mt   = blockIdx.x * 4 + wave;     // 0..127 (32-row tile)
    const int nb   = blockIdx.y * 64;
    const int n16  = lane & 15;
    const int hi8  = (lane & 16) ? 8 : 0;

    const int row0 = mt * 32;
    const bf16* arow0 = At + (size_t)(row0 + n16) * EE;
    const bf16* arow1 = At + (size_t)(row0 + 16 + n16) * EE;

    v8f acc[2][4] = {};
    for (int kk = 0; kk < EE; kk += 32) {
        v16bf a0 = afrag_bf16(arow0, kk, lane);
        v16bf a1 = afrag_bf16(arow1, kk, lane);
#pragma unroll
        for (int t = 0; t < 4; ++t) {
            const bf16* brow = Wo + (size_t)(nb + t * 16 + n16) * EE;
            v16bf b = bfrag_bf16(brow, kk, lane);
            acc[0][t] = wmma_bf16(a0, b, acc[0][t]);
            acc[1][t] = wmma_bf16(a1, b, acc[1][t]);
        }
    }
#pragma unroll
    for (int t = 0; t < 4; ++t) {
        const int col = nb + t * 16 + n16;
        const float bias = bo[col];
#pragma unroll
        for (int mi = 0; mi < 2; ++mi) {
#pragma unroll
            for (int r = 0; r < 8; ++r) {
                const size_t m = (size_t)row0 + mi * 16 + r + hi8;
                Out[m * EE + col] = acc[mi][t][r] + bias;
            }
        }
    }
}

// ---------------------------------------------------------------------------
extern "C" void kernel_launch(void* const* d_in, const int* in_sizes, int n_in,
                              void* d_out, int out_size, void* d_ws, size_t ws_size,
                              hipStream_t stream) {
    // setup_inputs order: key, query, value, Wq, bq, Wk, bk, Wv, bv, Wo, bo
    const float* key   = (const float*)d_in[0];
    const float* query = (const float*)d_in[1];
    const float* value = (const float*)d_in[2];
    const float* Wq = (const float*)d_in[3];
    const float* bq = (const float*)d_in[4];
    const float* Wk = (const float*)d_in[5];
    const float* bk = (const float*)d_in[6];
    const float* Wv = (const float*)d_in[7];
    const float* bv = (const float*)d_in[8];
    const float* Wo = (const float*)d_in[9];
    const float* bo = (const float*)d_in[10];
    float* out = (float*)d_out;

    constexpr size_t NX = (size_t)ROWS * EE;   // 4M elems (act / intermediates)
    constexpr size_t NW = (size_t)EE * EE;     // 1M elems (weights)

    // Workspace carve (bf16): 4*NX intermediates + 3*NX acts + 4*NW weights
    bf16* Qh  = (bf16*)d_ws;        // [B,H,S,Dh]
    bf16* Kh  = Qh  + NX;           // [B,H,S,Dh]
    bf16* Vt  = Kh  + NX;           // [B,H,Dh,S]
    bf16* At  = Vt  + NX;           // [B*S, E]
    bf16* Xqb = At  + NX;           // query bf16
    bf16* Xkb = Xqb + NX;           // key   bf16
    bf16* Xvb = Xkb + NX;           // value bf16
    bf16* Wqb = Xvb + NX;
    bf16* Wkb = Wqb + NW;
    bf16* Wvb = Wkb + NW;
    bf16* Wob = Wvb + NW;

    // ---- stage 0: convert fp32 inputs to bf16 (once per call)
    const int thr = 256;
    cvt_bf16_kernel<<<2048, thr, 0, stream>>>(query, Xqb, (int)(NX / 8));
    cvt_bf16_kernel<<<2048, thr, 0, stream>>>(key,   Xkb, (int)(NX / 8));
    cvt_bf16_kernel<<<2048, thr, 0, stream>>>(value, Xvb, (int)(NX / 8));
    cvt_bf16_kernel<<<512,  thr, 0, stream>>>(Wq, Wqb, (int)(NW / 8));
    cvt_bf16_kernel<<<512,  thr, 0, stream>>>(Wk, Wkb, (int)(NW / 8));
    cvt_bf16_kernel<<<512,  thr, 0, stream>>>(Wv, Wvb, (int)(NW / 8));
    cvt_bf16_kernel<<<512,  thr, 0, stream>>>(Wo, Wob, (int)(NW / 8));

    // ---- stage 1..3
    qkv_proj_kernel<<<dim3(32, 16, 3), 128, 0, stream>>>(
        Xqb, Xkb, Xvb, Wqb, Wkb, Wvb, bq, bk, bv, Qh, Kh, Vt);
    attention_kernel<<<dim3(1024, 1, 1), 128, 0, stream>>>(Qh, Kh, Vt, At);
    out_proj_kernel<<<dim3(32, 16, 1), 128, 0, stream>>>(At, Wob, bo, out);
}
